// VectorQuantizer_87041807220989
// MI455X (gfx1250) — matrix operand order repaired and checked by reading, not compiled, not run on hardware
//
#include <hip/hip_runtime.h>

#define D_DIM 768
#define K_CODES 8192
#define B_ROWS 32768
#define N_SLICE 4
#define WAVES_PER_WG 8
#define M_WAVE 32
#define M_WG (WAVES_PER_WG * M_WAVE)   /* 256 rows per workgroup */
#define N_TILE 64
#define COMMIT 0.25f

typedef __attribute__((ext_vector_type(16))) __bf16        v16bf;
typedef __attribute__((ext_vector_type(8)))  float         v8f;
typedef __attribute__((ext_vector_type(8)))  unsigned int  v8u;

// Build a 16-element bf16 fragment from two 8-element (16B) pre-packed runs.
static __device__ inline v16bf frag_from(const unsigned short* __restrict__ p, int e1) {
  uint4 a = *(const uint4*)(p);
  uint4 b = *(const uint4*)(p + e1);
  v8u u = { a.x, a.y, a.z, a.w, b.x, b.y, b.z, b.w };
  return __builtin_bit_cast(v16bf, u);
}

// ---------- one-time fp32 -> (hi, lo) bf16 split (truncation split) ----------
__global__ __launch_bounds__(256) void vq_split_kernel(
    const float* __restrict__ src, unsigned short* __restrict__ hi,
    unsigned short* __restrict__ lo, int n4) {
  const int i = blockIdx.x * 256 + threadIdx.x;
  if (i >= n4) return;
  float4 f = ((const float4*)src)[i];
  unsigned u0 = __float_as_uint(f.x), u1 = __float_as_uint(f.y);
  unsigned u2 = __float_as_uint(f.z), u3 = __float_as_uint(f.w);
  float h0 = __uint_as_float(u0 & 0xffff0000u), h1 = __uint_as_float(u1 & 0xffff0000u);
  float h2 = __uint_as_float(u2 & 0xffff0000u), h3 = __uint_as_float(u3 & 0xffff0000u);
  unsigned l0 = __float_as_uint(f.x - h0) >> 16, l1 = __float_as_uint(f.y - h1) >> 16;
  unsigned l2 = __float_as_uint(f.z - h2) >> 16, l3 = __float_as_uint(f.w - h3) >> 16;
  ((uint2*)hi)[i] = make_uint2((u0 >> 16) | (u1 & 0xffff0000u),
                               (u2 >> 16) | (u3 & 0xffff0000u));
  ((uint2*)lo)[i] = make_uint2(l0 | (l1 << 16), l2 | (l3 << 16));
}

// ---------- codebook row norms ----------
__global__ __launch_bounds__(256) void vq_cnorm_kernel(const float* __restrict__ cb,
                                                       float* __restrict__ cnorm) {
  const int lane = threadIdx.x & 31;
  const int row  = blockIdx.x * 8 + (threadIdx.x >> 5);
  const float* p = cb + (size_t)row * D_DIM;
  float s = 0.f;
  for (int i = lane; i < D_DIM; i += 32) { float v = p[i]; s = fmaf(v, v, s); }
#pragma unroll
  for (int m = 1; m < 32; m <<= 1) s += __shfl_xor(s, m, 32);
  if (lane == 0) cnorm[row] = s;
}

// ---------- fused GEMM (pre-split bf16 WMMA) + running argmin ----------
__global__ __launch_bounds__(256) void vq_argmin_kernel(
    const unsigned short* __restrict__ xhi, const unsigned short* __restrict__ xlo,
    const unsigned short* __restrict__ chi, const unsigned short* __restrict__ clo,
    const float* __restrict__ cnorm,
    float* __restrict__ pVal, int* __restrict__ pIdx) {
  const int lane = threadIdx.x & 31;
  const int wave = threadIdx.x >> 5;
  const int r    = lane & 15;
  const int half = lane >> 4;
  const int kbA  = half * 8;    // A frag: K = kbA+0..7 and 16+kbA+0..7 (ISA 7.12.2)
  const int kbB  = half * 8;    // B frag: K = kbB+0..7 and 16+kbB+0..7 (matches A)
  const int m0   = blockIdx.x * M_WG + wave * M_WAVE;
  const int nBeg = blockIdx.y * (K_CODES / N_SLICE);
  const int nEnd = nBeg + (K_CODES / N_SLICE);

  const unsigned short* a0h = xhi + (size_t)(m0 + r)      * D_DIM + kbA;
  const unsigned short* a0l = xlo + (size_t)(m0 + r)      * D_DIM + kbA;
  const unsigned short* a1h = xhi + (size_t)(m0 + 16 + r) * D_DIM + kbA;
  const unsigned short* a1l = xlo + (size_t)(m0 + 16 + r) * D_DIM + kbA;

  float bestVal[2][8];
  int   bestIdx[2][8];
#pragma unroll
  for (int g = 0; g < 2; ++g)
#pragma unroll
    for (int rr = 0; rr < 8; ++rr) { bestVal[g][rr] = __builtin_inff(); bestIdx[g][rr] = 0x7fffffff; }

  for (int n0 = nBeg; n0 < nEnd; n0 += N_TILE) {
    v8f acc[2][4];
#pragma unroll
    for (int g = 0; g < 2; ++g)
#pragma unroll
      for (int j = 0; j < 4; ++j) acc[g][j] = (v8f){};

    const unsigned short* bh[4];
    const unsigned short* bl[4];
#pragma unroll
    for (int j = 0; j < 4; ++j) {
      const size_t off = (size_t)(n0 + j * 16 + r) * D_DIM + kbB;
      bh[j] = chi + off;
      bl[j] = clo + off;
    }

    for (int d0 = 0; d0 < D_DIM; d0 += 32) {
      v16bf aHi[2], aLo[2], bHi[4], bLo[4];
      aHi[0] = frag_from(a0h + d0, 16);
      aLo[0] = frag_from(a0l + d0, 16);
      aHi[1] = frag_from(a1h + d0, 16);
      aLo[1] = frag_from(a1l + d0, 16);
#pragma unroll
      for (int j = 0; j < 4; ++j) {
        bHi[j] = frag_from(bh[j] + d0, 16);
        bLo[j] = frag_from(bl[j] + d0, 16);
      }
#pragma unroll
      for (int g = 0; g < 2; ++g)
#pragma unroll
        for (int j = 0; j < 4; ++j) {
          acc[g][j] = __builtin_amdgcn_wmma_f32_16x16x32_bf16(
              false, aHi[g], false, bHi[j], (short)0, acc[g][j], false, false);
          acc[g][j] = __builtin_amdgcn_wmma_f32_16x16x32_bf16(
              false, aHi[g], false, bLo[j], (short)0, acc[g][j], false, false);
          acc[g][j] = __builtin_amdgcn_wmma_f32_16x16x32_bf16(
              false, aLo[g], false, bHi[j], (short)0, acc[g][j], false, false);
        }
    }

    // distance = ||e||^2 - 2*dot  (||x||^2 constant per row, argmin-invariant)
#pragma unroll
    for (int j = 0; j < 4; ++j) {
      const int   n  = n0 + j * 16 + r;
      const float cn = cnorm[n];
#pragma unroll
      for (int g = 0; g < 2; ++g)
#pragma unroll
        for (int rr = 0; rr < 8; ++rr) {
          float dist = fmaf(-2.0f, acc[g][j][rr], cn);
          if (dist < bestVal[g][rr]) { bestVal[g][rr] = dist; bestIdx[g][rr] = n; }
        }
    }
  }

  // reduce the 16 N-candidate lanes per row (xor within each 16-lane half)
#pragma unroll
  for (int g = 0; g < 2; ++g)
#pragma unroll
    for (int rr = 0; rr < 8; ++rr) {
      float v  = bestVal[g][rr];
      int   id = bestIdx[g][rr];
#pragma unroll
      for (int m = 1; m < 16; m <<= 1) {
        float ov = __shfl_xor(v, m, 32);
        int   oi = __shfl_xor(id, m, 32);
        if (ov < v || (ov == v && oi < id)) { v = ov; id = oi; }
      }
      if (r == 0) {
        const int row = m0 + g * 16 + rr + 8 * half;   // C/D layout: M = rr + 8*half
        pVal[(size_t)row * N_SLICE + blockIdx.y] = v;
        pIdx[(size_t)row * N_SLICE + blockIdx.y] = id;
      }
    }
}

// ---------- combine K-slices -> final indices (raw int32 into d_out) ----------
__global__ __launch_bounds__(256) void vq_combine_kernel(
    const float* __restrict__ pVal, const int* __restrict__ pIdx,
    int* __restrict__ idxOut) {
  const int row = blockIdx.x * blockDim.x + threadIdx.x;
  float bv = __builtin_inff();
  int   bi = 0x7fffffff;
#pragma unroll
  for (int s = 0; s < N_SLICE; ++s) {
    float v = pVal[(size_t)row * N_SLICE + s];
    int   i = pIdx[(size_t)row * N_SLICE + s];
    if (v < bv || (v == bv && i < bi)) { bv = v; bi = i; }
  }
  idxOut[row] = bi;
}

// ---------- gather quantized + loss = (1+0.25)*mean((q-x)^2) ----------
__global__ __launch_bounds__(256) void vq_gather_loss_kernel(
    const float* __restrict__ x, const float* __restrict__ cb,
    const int* __restrict__ idx, float* __restrict__ qout,
    float* __restrict__ lossOut) {
  const int row  = blockIdx.x;
  const int code = idx[row];
  const float* src = cb + (size_t)code * D_DIM;
  const float* xr  = x  + (size_t)row  * D_DIM;
  float s = 0.f;
  for (int i = threadIdx.x; i < D_DIM; i += 256) {
    float q = src[i];
    qout[(size_t)row * D_DIM + i] = q;   // quantized_st forward value == quantized
    float df = q - xr[i];
    s = fmaf(df, df, s);
  }
#pragma unroll
  for (int m = 1; m < 32; m <<= 1) s += __shfl_xor(s, m, 32);
  __shared__ float red[WAVES_PER_WG];
  const int lane = threadIdx.x & 31, wv = threadIdx.x >> 5;
  if (lane == 0) red[wv] = s;
  __syncthreads();
  if (threadIdx.x == 0) {
    float t = 0.f;
#pragma unroll
    for (int w = 0; w < WAVES_PER_WG; ++w) t += red[w];
    atomicAdd(lossOut, t * ((1.0f + COMMIT) / ((float)B_ROWS * (float)D_DIM)));
  }
}

extern "C" void kernel_launch(void* const* d_in, const int* in_sizes, int n_in,
                              void* d_out, int out_size, void* d_ws, size_t ws_size,
                              hipStream_t stream) {
  const float* x  = (const float*)d_in[0];
  // d_in[1] ("var") is unused by the reference
  const float* cb = (const float*)d_in[2];

  float* qout    = (float*)d_out;                                  // [B, D]
  float* lossOut = (float*)d_out + (size_t)B_ROWS * D_DIM;         // scalar
  int*   idxOut  = (int*)d_out + (size_t)B_ROWS * D_DIM + 1;       // [B] int32 bits

  // Stage split X inside the qout region (B*D*4 bytes == Xhi + Xlo exactly);
  // the gather kernel overwrites it last and reads only fp32 inputs.
  unsigned short* xhi = (unsigned short*)d_out;
  unsigned short* xlo = xhi + (size_t)B_ROWS * D_DIM;

  char*  ws    = (char*)d_ws;
  float* cnorm = (float*)ws;                                          // K floats
  float* pVal  = (float*)(ws + (size_t)K_CODES * 4);                  // B*N_SLICE floats
  int*   pIdx  = (int*)(ws + (size_t)K_CODES * 4
                           + (size_t)B_ROWS * N_SLICE * 4);           // B*N_SLICE ints
  unsigned short* chi = (unsigned short*)(ws + (size_t)K_CODES * 4
                           + (size_t)B_ROWS * N_SLICE * 8);           // K*D bf16
  unsigned short* clo = chi + (size_t)K_CODES * D_DIM;                // K*D bf16

  const int nx4 = (B_ROWS * D_DIM) / 4;
  const int nc4 = (K_CODES * D_DIM) / 4;
  vq_split_kernel<<<nx4 / 256, 256, 0, stream>>>(x,  xhi, xlo, nx4);
  vq_split_kernel<<<nc4 / 256, 256, 0, stream>>>(cb, chi, clo, nc4);

  vq_cnorm_kernel<<<K_CODES / 8, 256, 0, stream>>>(cb, cnorm);

  dim3 grid(B_ROWS / M_WG, N_SLICE);
  vq_argmin_kernel<<<grid, 256, 0, stream>>>(xhi, xlo, chi, clo, cnorm, pVal, pIdx);

  vq_combine_kernel<<<B_ROWS / 256, 256, 0, stream>>>(pVal, pIdx, idxOut);

  hipMemsetAsync(lossOut, 0, sizeof(float), stream);
  vq_gather_loss_kernel<<<B_ROWS, 256, 0, stream>>>(x, cb, idxOut, qout, lossOut);
}
